// GATPredictor_73272142069971
// MI455X (gfx1250) — matrix operand order repaired and checked by reading, not compiled, not run on hardware
//
#include <hip/hip_runtime.h>
#include <hip/hip_bf16.h>

// Problem constants (match reference)
#define NND 50000
#define NE  400000
#define NB  128
#define NH  4
#define NF  64
#define HFD 256     // H*F
#define NHID 512
#define EPSB 1e-5f

typedef __bf16 bf16;
typedef bf16  v16bf __attribute__((ext_vector_type(16)));
typedef float v8f   __attribute__((ext_vector_type(8)));

// ---------------------------------------------------------------------------
// Weight swizzle: W[K][Ncols] f32 -> bf16 in exact WMMA B-fragment order.
// Wsw[((cb*KS + ks)*32 + lane)*16 + j] = W[ks*32 + (lane>>4)*16 + j][cb*16 + (lane&15)]
// (B 32x16 order used here: lane holds column n=lane%16; lanes 0-15 K=0..15, 16-31 K=16..31,
//  K pairs contiguous per VGPR -> element j maps to K = half*16 + j.)
// ---------------------------------------------------------------------------
__global__ void w_swizzle_k(const float* __restrict__ W, bf16* __restrict__ Wsw,
                            int K, int Ncols) {
    int i = blockIdx.x * blockDim.x + threadIdx.x;
    if (i >= K * Ncols) return;
    int j    = i & 15;
    int lane = (i >> 4) & 31;
    int t    = i >> 9;          // cb*KS + ks
    int KS   = K >> 5;
    int ks   = t % KS;
    int cb   = t / KS;
    int k = ks * 32 + (lane >> 4) * 16 + j;
    int n = cb * 16 + (lane & 15);
    Wsw[i] = (bf16)W[(size_t)k * Ncols + n];
}

// ---------------------------------------------------------------------------
// WMMA GEMM: Out[M x Ncols] = A[M x K] * W[K x Ncols] (+bias, optional relu)
// A is f32 row-major; W is pre-swizzled bf16 (w_swizzle_k).
// Block = 256 threads = 8 waves; tile = 16 rows x 128 cols; grid = (M/16, Ncols/128).
// B fragments: 2 x global_load_b128 per lane straight from swizzled weights.
// A tile: staged into LDS in per-lane fragment order (1 KB), read as 2 x ds b128.
// ---------------------------------------------------------------------------
__global__ __launch_bounds__(256) void gemm_wmma_k(
    const float* __restrict__ A, const bf16* __restrict__ Wsw,
    float* __restrict__ Out, int K, int Ncols,
    const float* __restrict__ bias, int relu)
{
    __shared__ bf16 lAf[32][16];   // A tile, fragment-ordered (lane-major), 1 KB

    const int tid  = threadIdx.x;
    const int row0 = blockIdx.x * 16;
    const int col0 = blockIdx.y * 128;
    const int lane = tid & 31;
    const int wave = tid >> 5;             // 0..7 -> 16-col sub-tile
    const int l    = lane & 15;
    const int half = lane >> 4;
    const int KS   = K >> 5;
    const int cb   = (col0 >> 4) + wave;   // global 16-col tile index

    // Staging role of this thread: one bf16-pair of the 16x32 A tile per k-step.
    // A-fragment layout: lane half h, row r hold K = half*16..+16 (contiguous).
    const int sr = tid >> 4;               // tile row 0..15
    const int sk = (tid & 15) * 2;         // even K offset 0..30
    const int dLane = sr + 16 * (sk >> 4);
    const int dElem = sk & 15;

    v8f acc = {0.f, 0.f, 0.f, 0.f, 0.f, 0.f, 0.f, 0.f};

    for (int ks = 0; ks < KS; ++ks) {
        // B fragment: 32 contiguous bytes per lane from swizzled weights
        const uint4* wp = (const uint4*)(Wsw + (((size_t)(cb * KS + ks) * 32 + lane) << 4));
        uint4 b0 = wp[0];
        uint4 b1 = wp[1];

        // A staging: one float2 -> packed bf16 pair -> LDS
        const float* ap = A + (size_t)(row0 + sr) * K + ks * 32 + sk;
        float a0 = ap[0];
        float a1 = ap[1];
        if (ks + 1 < KS) __builtin_prefetch(ap + 32, 0, 3);
        union { unsigned int u; bf16 h[2]; } pk;
        pk.h[0] = (bf16)a0;
        pk.h[1] = (bf16)a1;
        *(unsigned int*)&lAf[dLane][dElem] = pk.u;
        __syncthreads();

        union { v16bf v; uint4 q[2]; } afr, bfr;
        afr.q[0] = *(const uint4*)&lAf[lane][0];
        afr.q[1] = *(const uint4*)&lAf[lane][8];
        bfr.q[0] = b0;
        bfr.q[1] = b1;
        acc = __builtin_amdgcn_wmma_f32_16x16x32_bf16(
            false, afr.v, false, bfr.v, (short)0, acc, false, false);
        __syncthreads();
    }

    // C/D layout: VGPR r -> row r (lanes 0-15) / row r+8 (lanes 16-31), col = lane%16
    #pragma unroll
    for (int r = 0; r < 8; ++r) {
        int row = row0 + r + half * 8;
        int col = col0 + wave * 16 + l;
        float v = acc[r];
        if (bias) v += bias[col];
        if (relu) v = fmaxf(v, 0.f);
        Out[(size_t)row * Ncols + col] = v;
    }
}

// ---------------------------------------------------------------------------
// Helpers
// ---------------------------------------------------------------------------
__device__ __forceinline__ void atomicMaxF(float* addr, float val) {
    // sign-aware int-punned float max (works with -1e30 init)
    if (val >= 0.f) atomicMax((int*)addr, __float_as_int(val));
    else            atomicMin((unsigned int*)addr, __float_as_uint(val));
}

__global__ void fill_k(float* __restrict__ p, float v, int n) {
    int i = blockIdx.x * blockDim.x + threadIdx.x;
    if (i < n) p[i] = v;
}

// el[n,h] = sum_f hf[n,h,f]*al[h,f] ; er likewise
__global__ void attn_coeff_k(const float* __restrict__ hf,
                             const float* __restrict__ al,
                             const float* __restrict__ ar,
                             float* __restrict__ el, float* __restrict__ er) {
    int i = blockIdx.x * blockDim.x + threadIdx.x;   // n*NH + h
    if (i >= NND * NH) return;
    int h = i & (NH - 1), n = i >> 2;
    const float* row = hf + (size_t)n * HFD + h * NF;
    const float* pl = al + h * NF;
    const float* pr = ar + h * NF;
    float sl = 0.f, sr = 0.f;
    #pragma unroll 8
    for (int f = 0; f < NF; ++f) { float v = row[f]; sl += v * pl[f]; sr += v * pr[f]; }
    el[i] = sl; er[i] = sr;
}

__device__ __forceinline__ float leaky02(float x) { return x > 0.f ? x : 0.2f * x; }

__global__ void edge_max_k(const int* __restrict__ src, const int* __restrict__ dst,
                           const float* __restrict__ el, const float* __restrict__ er,
                           float* __restrict__ m) {
    int i = blockIdx.x * blockDim.x + threadIdx.x;   // e*NH + h
    if (i >= NE * NH) return;
    int h = i & (NH - 1), e = i >> 2;
    int d = dst[e];
    float v = leaky02(el[src[e] * NH + h] + er[d * NH + h]);
    atomicMaxF(&m[d * NH + h], v);
}

__global__ void edge_expsum_k(const int* __restrict__ src, const int* __restrict__ dst,
                              const float* __restrict__ el, const float* __restrict__ er,
                              const float* __restrict__ m, float* __restrict__ denom) {
    int i = blockIdx.x * blockDim.x + threadIdx.x;
    if (i >= NE * NH) return;
    int h = i & (NH - 1), e = i >> 2;
    int d = dst[e];
    float v = leaky02(el[src[e] * NH + h] + er[d * NH + h]);
    atomicAdd(&denom[d * NH + h], __expf(v - m[d * NH + h]));
}

// one wave per (edge, head); lanes cover f and f+32 (coalesced atomics)
__global__ __launch_bounds__(256) void edge_msg_k(
    const int* __restrict__ src, const int* __restrict__ dst,
    const float* __restrict__ el, const float* __restrict__ er,
    const float* __restrict__ m, const float* __restrict__ denom,
    const float* __restrict__ hf, float* __restrict__ acc) {
    int w = blockIdx.x * 8 + (threadIdx.x >> 5);
    if (w >= NE * NH) return;
    int lane = threadIdx.x & 31;
    int h = w & (NH - 1), e = w >> 2;
    int s = src[e], d = dst[e];
    float v = leaky02(el[s * NH + h] + er[d * NH + h]);
    float a = __expf(v - m[d * NH + h]) / denom[d * NH + h];
    const float* hrow = hf + (size_t)s * HFD + h * NF;
    float* arow = acc + (size_t)d * HFD + h * NF;
    atomicAdd(&arow[lane],      hrow[lane]      * a);
    atomicAdd(&arow[lane + 32], hrow[lane + 32] * a);
}

__global__ void elu_k(float* __restrict__ p, int n) {
    int i = blockIdx.x * blockDim.x + threadIdx.x;
    if (i < n) { float x = p[i]; p[i] = x > 0.f ? x : (__expf(x) - 1.f); }
}

// acc2 = h1 (identity residual) + b2
__global__ void acc2_init_k(const float* __restrict__ h1, const float* __restrict__ b2,
                            float* __restrict__ acc2, int n) {
    int i = blockIdx.x * blockDim.x + threadIdx.x;
    if (i < n) acc2[i] = h1[i] + b2[i & (HFD - 1)];
}

// mean over heads: D[n,f] = 0.25 * sum_h acc[n,h,f]
__global__ void mean_heads_k(const float* __restrict__ acc, float* __restrict__ hm) {
    int i = blockIdx.x * blockDim.x + threadIdx.x;   // n*NF + f
    if (i >= NND * NF) return;
    int f = i & (NF - 1), n = i >> 6;
    const float* row = acc + (size_t)n * HFD;
    hm[i] = 0.25f * (row[f] + row[NF + f] + row[2 * NF + f] + row[3 * NF + f]);
}

__global__ void gate_k(const float* __restrict__ hm, const float* __restrict__ Wa,
                       const float* __restrict__ ba, float* __restrict__ wg) {
    int n = blockIdx.x * blockDim.x + threadIdx.x;
    if (n >= NND) return;
    const float* row = hm + (size_t)n * NF;
    float s = ba[0];
    #pragma unroll 8
    for (int f = 0; f < NF; ++f) s += row[f] * Wa[f];
    wg[n] = 1.f / (1.f + __expf(-s));
}

__global__ void readout_k(const float* __restrict__ hm, const float* __restrict__ wg,
                          const int* __restrict__ n2g,
                          float* __restrict__ hsum, float* __restrict__ hmax) {
    int i = blockIdx.x * blockDim.x + threadIdx.x;   // n*NF + f
    if (i >= NND * NF) return;
    int f = i & (NF - 1), n = i >> 6;
    int g = n2g[n];
    float v = hm[i];
    atomicAdd(&hsum[g * NF + f], wg[n] * v);
    atomicMaxF(&hmax[g * NF + f], v);
}

__global__ void concat_k(const float* __restrict__ hsum, const float* __restrict__ hmax,
                         float* __restrict__ g) {
    int i = blockIdx.x * blockDim.x + threadIdx.x;   // b*128 + j
    if (i >= NB * 2 * NF) return;
    int j = i & (2 * NF - 1), b = i >> 7;
    g[i] = (j < NF) ? hsum[b * NF + j] : hmax[b * NF + (j - NF)];
}

__global__ void bn_stats_k(const float* __restrict__ z, float* __restrict__ mu,
                           float* __restrict__ var) {
    int j = blockIdx.x * blockDim.x + threadIdx.x;
    if (j >= NHID) return;
    float s = 0.f;
    for (int b = 0; b < NB; ++b) s += z[(size_t)b * NHID + j];
    float m = s / (float)NB;
    float q = 0.f;
    for (int b = 0; b < NB; ++b) { float d = z[(size_t)b * NHID + j] - m; q += d * d; }
    mu[j] = m; var[j] = q / (float)NB;
}

__global__ void final_k(const float* __restrict__ z, const float* __restrict__ mu,
                        const float* __restrict__ var, const float* __restrict__ gamma,
                        const float* __restrict__ beta, const float* __restrict__ Wm2,
                        const float* __restrict__ bm2, float* __restrict__ out) {
    int b = threadIdx.x;
    if (b >= NB) return;
    float s = bm2[0];
    for (int j = 0; j < NHID; ++j) {
        float zn = (z[(size_t)b * NHID + j] - mu[j]) * rsqrtf(var[j] + EPSB) * gamma[j] + beta[j];
        s += zn * Wm2[j];
    }
    out[b] = s;
}

// ---------------------------------------------------------------------------
extern "C" void kernel_launch(void* const* d_in, const int* in_sizes, int n_in,
                              void* d_out, int out_size, void* d_ws, size_t ws_size,
                              hipStream_t stream) {
    const float* feats = (const float*)d_in[0];
    const int*   src   = (const int*)d_in[1];
    const int*   dst   = (const int*)d_in[2];
    const int*   n2g   = (const int*)d_in[3];
    const float* W1    = (const float*)d_in[4];
    const float* al1   = (const float*)d_in[5];
    const float* ar1   = (const float*)d_in[6];
    const float* resW1 = (const float*)d_in[7];
    const float* b1    = (const float*)d_in[8];
    const float* W2    = (const float*)d_in[9];
    const float* al2   = (const float*)d_in[10];
    const float* ar2   = (const float*)d_in[11];
    const float* b2    = (const float*)d_in[12];
    const float* Wa    = (const float*)d_in[13];
    const float* ba    = (const float*)d_in[14];
    const float* Wm1   = (const float*)d_in[15];
    const float* bm1   = (const float*)d_in[16];
    const float* gamma = (const float*)d_in[17];
    const float* beta  = (const float*)d_in[18];
    const float* Wm2   = (const float*)d_in[19];
    const float* bm2   = (const float*)d_in[20];
    float* out = (float*)d_out;

    float* ws = (float*)d_ws;
    float* hf    = ws;                                 // N x 256   (hf1 then hf2)
    float* accB  = hf    + (size_t)NND * HFD;          // N x 256   (acc1 -> h1)
    float* accC  = accB  + (size_t)NND * HFD;          // N x 256   (acc2)
    float* hmean = accC  + (size_t)NND * HFD;          // N x 64
    float* el    = hmean + (size_t)NND * NF;           // N x 4
    float* er    = el    + (size_t)NND * NH;
    float* mbuf  = er    + (size_t)NND * NH;
    float* denom = mbuf  + (size_t)NND * NH;
    float* wg    = denom + (size_t)NND * NH;           // N
    float* hsum  = wg    + (size_t)NND;                // B x 64
    float* hmax  = hsum  + (size_t)NB * NF;
    float* gbuf  = hmax  + (size_t)NB * NF;            // B x 128
    float* zbuf  = gbuf  + (size_t)NB * 2 * NF;        // B x 512
    float* mu    = zbuf  + (size_t)NB * NHID;          // 512
    float* var   = mu + NHID;                          // 512
    // bf16 swizzled weights (32-byte aligned region)
    size_t foff = (size_t)((var + NHID) - ws);
    foff = (foff + 7) & ~(size_t)7;
    bf16* wsw1 = (bf16*)(ws + foff);                   // 128*256
    bf16* wswR = wsw1 + (size_t)128 * HFD;             // 128*256
    bf16* wsw2 = wswR + (size_t)128 * HFD;             // 256*256
    bf16* wswM = wsw2 + (size_t)HFD * HFD;             // 128*512

    dim3 blk(256);
    const int nhB = (NND * NH + 255) / 256;
    const int ehB = (NE * NH + 255) / 256;

    // ---- weight swizzle passes (bf16, WMMA fragment order) ----
    w_swizzle_k<<<(128 * HFD) / 256, blk, 0, stream>>>(W1,    wsw1, 128, HFD);
    w_swizzle_k<<<(128 * HFD) / 256, blk, 0, stream>>>(resW1, wswR, 128, HFD);
    w_swizzle_k<<<(HFD * HFD) / 256, blk, 0, stream>>>(W2,    wsw2, HFD, HFD);
    w_swizzle_k<<<(128 * NHID) / 256, blk, 0, stream>>>(Wm1,  wswM, 128, NHID);

    // ---- GAT layer 1 ----
    gemm_wmma_k<<<dim3(NND / 16, HFD / 128), blk, 0, stream>>>(feats, wsw1, hf, 128, HFD, nullptr, 0);
    gemm_wmma_k<<<dim3(NND / 16, HFD / 128), blk, 0, stream>>>(feats, wswR, accB, 128, HFD, b1, 0);
    attn_coeff_k<<<nhB, blk, 0, stream>>>(hf, al1, ar1, el, er);
    fill_k<<<nhB, blk, 0, stream>>>(mbuf, -1e30f, NND * NH);
    fill_k<<<nhB, blk, 0, stream>>>(denom, 0.f, NND * NH);
    edge_max_k<<<ehB, blk, 0, stream>>>(src, dst, el, er, mbuf);
    edge_expsum_k<<<ehB, blk, 0, stream>>>(src, dst, el, er, mbuf, denom);
    edge_msg_k<<<(NE * NH) / 8, blk, 0, stream>>>(src, dst, el, er, mbuf, denom, hf, accB);
    elu_k<<<(NND * HFD) / 256, blk, 0, stream>>>(accB, NND * HFD);   // accB = h1

    // ---- GAT layer 2 ----
    gemm_wmma_k<<<dim3(NND / 16, HFD / 128), blk, 0, stream>>>(accB, wsw2, hf, HFD, HFD, nullptr, 0);
    acc2_init_k<<<(NND * HFD) / 256, blk, 0, stream>>>(accB, b2, accC, NND * HFD);
    attn_coeff_k<<<nhB, blk, 0, stream>>>(hf, al2, ar2, el, er);
    fill_k<<<nhB, blk, 0, stream>>>(mbuf, -1e30f, NND * NH);
    fill_k<<<nhB, blk, 0, stream>>>(denom, 0.f, NND * NH);
    edge_max_k<<<ehB, blk, 0, stream>>>(src, dst, el, er, mbuf);
    edge_expsum_k<<<ehB, blk, 0, stream>>>(src, dst, el, er, mbuf, denom);
    edge_msg_k<<<(NE * NH) / 8, blk, 0, stream>>>(src, dst, el, er, mbuf, denom, hf, accC);
    mean_heads_k<<<(NND * NF) / 256, blk, 0, stream>>>(accC, hmean);

    // ---- readout ----
    gate_k<<<(NND + 255) / 256, blk, 0, stream>>>(hmean, Wa, ba, wg);
    fill_k<<<(NB * NF + 255) / 256, blk, 0, stream>>>(hsum, 0.f, NB * NF);
    fill_k<<<(NB * NF + 255) / 256, blk, 0, stream>>>(hmax, -1e30f, NB * NF);
    readout_k<<<(NND * NF) / 256, blk, 0, stream>>>(hmean, wg, n2g, hsum, hmax);
    concat_k<<<(NB * 2 * NF) / 256, blk, 0, stream>>>(hsum, hmax, gbuf);

    // ---- MLP head ----
    gemm_wmma_k<<<dim3(NB / 16, NHID / 128), blk, 0, stream>>>(gbuf, wswM, zbuf, 2 * NF, NHID, bm1, 1);
    bn_stats_k<<<2, blk, 0, stream>>>(zbuf, mu, var);
    final_k<<<1, 128, 0, stream>>>(zbuf, mu, var, gamma, beta, Wm2, bm2, out);
}
